// QNetwork_12859132084781
// MI455X (gfx1250) — matrix operand order repaired and checked by reading, not compiled, not run on hardware
//
#include <hip/hip_runtime.h>

// ---------------------------------------------------------------------------
// Fused Q-network forward for MI455X (gfx1250, wave32, WMMA).
//   B=16384 samples, N=32 neighbours, dims 8 -> 256 -> 256 -> 128, then
//   mask-pool + action-select -> 259 -> 256 -> 1.
//
// Compute ~108 GFLOP vs ~11 MB HBM traffic => WMMA-bound, but L2->WGP weight
// streaming is the secondary ceiling. This version:
//   * one sample (32 rows, 2 M-tiles) per wave: each B fragment feeds 2 WMMAs
//   * A fragments held in registers for layers 2/3 (loaded once per layer)
//   * layer-1 A fragments built straight from x_feat (K zero-padded 8->32)
//   * all activation LDS traffic is wave-private -> single barrier pre-pool
// ---------------------------------------------------------------------------

typedef __attribute__((ext_vector_type(16))) _Float16 v16h;
typedef __attribute__((ext_vector_type(8)))  _Float16 v8h;
typedef __attribute__((ext_vector_type(8)))  float    v8f;

#define BATCH   16384
#define NNEI    32
#define DIN     8
#define EDIM    128
#define FDIM    256
#define XFEAT   163        // 3 + 5*32
#define FCIN    (2*EDIM+3) // 259
#define EPSV    1e-5f
#define SLOPE   0.01f

// workspace layout (in _Float16 units): f16 weights, row-major [outF][K]
#define W1H_OFF 0                       // [256][32]  (K zero-padded 8 -> 32)
#define W2H_OFF (256*32)                // [256][256]
#define W3H_OFF (W2H_OFF + 256*256)     // [128][256]
#define WS_HALVES (W3H_OFF + 128*256)   // 106496 halves = 208 KB

// ---------------------------------------------------------------------------
// Kernel 0: convert f32 weights to f16 row-major (B-fragment friendly layout).
// ---------------------------------------------------------------------------
__global__ void prep_weights(const float* __restrict__ W1,
                             const float* __restrict__ W2,
                             const float* __restrict__ W3,
                             _Float16* __restrict__ wsH) {
  int i = blockIdx.x * blockDim.x + threadIdx.x;
  if (i < 256 * 32) {
    int f = i >> 5, k = i & 31;
    wsH[W1H_OFF + i] = (k < DIN) ? (_Float16)W1[f * DIN + k] : (_Float16)0.0f;
  } else if (i < 256 * 32 + 256 * 256) {
    int j = i - 256 * 32;
    wsH[W2H_OFF + j] = (_Float16)W2[j];
  } else if (i < WS_HALVES) {
    int j = i - (256 * 32 + 256 * 256);
    wsH[W3H_OFF + j] = (_Float16)W3[j];
  }
}

// ---------------------------------------------------------------------------
// Fragment loaders (wave32, 16-bit WMMA layouts, CDNA5 ISA 7.12.2).
//
// A (16x32, MxK) from row-major halves with row stride `stride`:
//   lane L holds row M = L%16; K = {0..7,16..23} (L<16) or {8..15,24..31}.
// B (32x16, KxN) from row-major weight storage W[f][k] (f = output feature):
//   lane L holds column N = L%16 -> f = fbase + L%16,
//   K = {0..7,16..23} (L<16) or {8..15,24..31} (L>=16), same as A.
// ---------------------------------------------------------------------------
__device__ __forceinline__ v16h load_a_frag(const _Float16* base, int stride,
                                            int k0, int lane) {
  int m = lane & 15;
  int koff = (lane < 16) ? 0 : 8;
  const _Float16* p = base + m * stride + k0 + koff;
  v8h lo = *(const v8h*)(p);       // K = k0+koff .. +7
  v8h hi = *(const v8h*)(p + 16);  // K = k0+koff+16 .. +23
  v16h r;
#pragma unroll
  for (int i = 0; i < 8; ++i) { r[i] = lo[i]; r[8 + i] = hi[i]; }
  return r;
}

__device__ __forceinline__ v16h load_b_frag(const _Float16* W, int stride,
                                            int fbase, int k0, int lane) {
  int n = lane & 15;
  int koff = (lane < 16) ? 0 : 8;
  const _Float16* p = W + (size_t)(fbase + n) * stride + k0 + koff;
  v8h lo = *(const v8h*)(p);
  v8h hi = *(const v8h*)(p + 16);
  v16h r;
#pragma unroll
  for (int i = 0; i < 8; ++i) { r[i] = lo[i]; r[8 + i] = hi[i]; }
  return r;
}

// Load all A fragments for a 32-row / 256-col activation block into registers.
__device__ __forceinline__ void load_a_frags8(const _Float16* base, int lane,
                                              v16h* aF) {
#pragma unroll
  for (int mt = 0; mt < 2; ++mt)
#pragma unroll
    for (int kc = 0; kc < 8; ++kc)
      aF[mt * 8 + kc] = load_a_frag(base + mt * 16 * 256, 256, kc * 32, lane);
}

// ---------------------------------------------------------------------------
// One MLP layer for a wave's 32-row block (2 M-tiles), A in registers:
//   out = A @ W^T + bias (WMMA f32 acc), optionally LayerNorm + LeakyReLU.
// D layout: VGPR g -> row M = g + (lane<16 ? 0 : 8), col N = lane&15.
// Row stats reduced over 16-lane halves with shfl_xor(1,2,4,8); pre-LN values
// staged to wave-private LDS (f16), LN applied in a second pass (stats f32).
// ---------------------------------------------------------------------------
template <int NK, int NT, bool LN>
__device__ __forceinline__ void layer_regA(
    const v16h* aF,                                   // [2*NK] A fragments
    const _Float16* __restrict__ Wh, int wStride,     // f16 weights [F][K]
    const float* __restrict__ bias,
    const float* __restrict__ gamma,
    const float* __restrict__ beta,
    _Float16* outBase, int outStride,                 // wave-private block
    int lane) {
  const int n = lane & 15;
  const int rowOff = (lane < 16) ? 0 : 8;
  float rsum[2][8], rsq[2][8];
  if constexpr (LN) {
#pragma unroll
    for (int mt = 0; mt < 2; ++mt)
#pragma unroll
      for (int g = 0; g < 8; ++g) { rsum[mt][g] = 0.f; rsq[mt][g] = 0.f; }
  }

  for (int nt = 0; nt < NT; ++nt) {
    v8f acc[2];
#pragma unroll
    for (int mt = 0; mt < 2; ++mt)
      acc[mt] = (v8f){0.f, 0.f, 0.f, 0.f, 0.f, 0.f, 0.f, 0.f};
#pragma unroll
    for (int kc = 0; kc < NK; ++kc) {
      v16h b = load_b_frag(Wh, wStride, nt * 16, kc * 32, lane);
#pragma unroll
      for (int mt = 0; mt < 2; ++mt)
        acc[mt] = __builtin_amdgcn_wmma_f32_16x16x32_f16(
            false, aF[mt * NK + kc], false, b, (short)0, acc[mt], false, false);
    }
    float bv = bias[nt * 16 + n];
#pragma unroll
    for (int mt = 0; mt < 2; ++mt)
#pragma unroll
      for (int g = 0; g < 8; ++g) {
        float v = acc[mt][g] + bv;
        if constexpr (LN) { rsum[mt][g] += v; rsq[mt][g] += v * v; }
        outBase[(mt * 16 + g + rowOff) * outStride + nt * 16 + n] = (_Float16)v;
      }
  }

  if constexpr (LN) {
    float mean[2][8], rstd[2][8];
    const float invD = 1.0f / (float)(NT * 16);
#pragma unroll
    for (int mt = 0; mt < 2; ++mt)
#pragma unroll
      for (int g = 0; g < 8; ++g) {
#pragma unroll
        for (int m = 1; m < 16; m <<= 1) {
          rsum[mt][g] += __shfl_xor(rsum[mt][g], m, 32);
          rsq[mt][g]  += __shfl_xor(rsq[mt][g],  m, 32);
        }
        mean[mt][g] = rsum[mt][g] * invD;
        float var = rsq[mt][g] * invD - mean[mt][g] * mean[mt][g];
        rstd[mt][g] = rsqrtf(var + EPSV);
      }
    for (int nt = 0; nt < NT; ++nt) {
      float gw = gamma[nt * 16 + n];
      float bw = beta[nt * 16 + n];
#pragma unroll
      for (int mt = 0; mt < 2; ++mt)
#pragma unroll
        for (int g = 0; g < 8; ++g) {
          int idx = (mt * 16 + g + rowOff) * outStride + nt * 16 + n;
          float v = (float)outBase[idx];
          v = (v - mean[mt][g]) * rstd[mt][g] * gw + bw;
          v = (v >= 0.f) ? v : SLOPE * v;  // leaky relu
          outBase[idx] = (_Float16)v;
        }
    }
  }
}

// ---------------------------------------------------------------------------
// Kernel 1: 64 threads = 2 waves; each wave owns one sample (32 rows).
// ---------------------------------------------------------------------------
__global__ __launch_bounds__(64) void qnet_kernel(
    const float* __restrict__ x_feat,
    const float* __restrict__ b1, const float* __restrict__ g1, const float* __restrict__ be1,
    const float* __restrict__ b2, const float* __restrict__ g2, const float* __restrict__ be2,
    const float* __restrict__ b3,
    const float* __restrict__ Wf1, const float* __restrict__ bf1,
    const float* __restrict__ Wf2, const float* __restrict__ bf2,
    const int* __restrict__ actions,
    const _Float16* __restrict__ wsH,
    float* __restrict__ out) {
  __shared__ __align__(16) _Float16 bufAct[2 * NNEI * 256];  // 32 KB, per-wave 16 KB
  __shared__ float s_mask[2 * NNEI];
  __shared__ float s_fc[2][272];   // [cur(3)|selected(128)|pooled(128)]
  __shared__ float s_q[2];

  const int blk  = blockIdx.x;
  const int t    = threadIdx.x;
  const int lane = t & 31;
  const int w    = t >> 5;                       // wave id == local sample id
  const int smp  = blk * 2 + w;                  // global sample
  const float* xrow = x_feat + (size_t)smp * XFEAT;
  _Float16* actBase = bufAct + w * (NNEI * 256); // wave-private region

  if (t < 2) s_q[t] = 0.f;

  const _Float16* w1h = wsH + W1H_OFF;
  const _Float16* w2h = wsH + W2H_OFF;
  const _Float16* w3h = wsH + W3H_OFF;

  // ---- layer-1 A fragments straight from x_feat (registers, no staging) ---
  // Lane L<16 holds row m = L (K=0..7 real, 16..23 pad); lanes >=16 all pad.
  v16h a1[2];
#pragma unroll
  for (int mt = 0; mt < 2; ++mt) {
    v16h a;
#pragma unroll
    for (int i = 0; i < 16; ++i) a[i] = (_Float16)0.f;
    if (lane < 16) {
      int nrow = mt * 16 + lane;
      a[0] = (_Float16)xrow[0];
      a[1] = (_Float16)xrow[1];
      a[2] = (_Float16)xrow[2];
      bool any = false;
#pragma unroll
      for (int j = 0; j < 5; ++j) {
        float v = xrow[3 + 5 * nrow + j];
        any = any || (v != 0.f);
        a[3 + j] = (_Float16)v;
      }
      s_mask[w * NNEI + nrow] = any ? 1.f : 0.f;
    }
    a1[mt] = a;
  }

  // ---- layer1: regs -> actBase (stride 256), LN + leaky -------------------
  layer_regA<1, 16, true>(a1, w1h, 32, b1, g1, be1, actBase, 256, lane);

  // ---- layer2: hoist A to regs, actBase -> actBase ------------------------
  v16h aF[16];
  load_a_frags8(actBase, lane, aF);
  layer_regA<8, 16, true>(aF, w2h, 256, b2, g2, be2, actBase, 256, lane);

  // ---- layer3: hoist A to regs, embeds into actBase (stride 128, no LN) ---
  load_a_frags8(actBase, lane, aF);
  layer_regA<8, 8, false>(aF, w3h, 256, b3, nullptr, nullptr, actBase, 128, lane);

  __syncthreads();  // masks + embeds visible block-wide

  // ---- masked mean-pool + action-select + cur_pos -------------------------
  for (int job = t; job < 2 * EDIM; job += 64) {
    int s = job >> 7;         // local sample
    int c = job & 127;        // embed column
    const _Float16* emb = bufAct + s * (NNEI * 256);  // stride 128 region
    float sum = 0.f, cnt = 0.f;
    for (int nr = 0; nr < NNEI; ++nr) {
      float m = s_mask[s * NNEI + nr];
      sum += m * (float)emb[nr * 128 + c];
      cnt += m;
    }
    float denom = (cnt < 1.f) ? 1.f : cnt;
    int act = actions[blk * 2 + s];
    s_fc[s][3 + c]        = (float)emb[act * 128 + c];  // selected
    s_fc[s][3 + EDIM + c] = sum / denom;                // pooled
  }
  if (t < 6) {
    int s = t / 3;
    s_fc[s][t % 3] = x_feat[(size_t)(blk * 2 + s) * XFEAT + (t % 3)];
  }
  __syncthreads();

  // ---- final FC: 259 -> 256 (leaky) -> 1 for both samples -----------------
  float qp0 = 0.f, qp1 = 0.f;
#pragma unroll
  for (int j = 0; j < 4; ++j) {
    int f = t + 64 * j;
    const float* wrow = Wf1 + (size_t)f * FCIN;
    float d0 = 0.f, d1 = 0.f;
    for (int k = 0; k < FCIN; ++k) {
      float wv = wrow[k];
      d0 += s_fc[0][k] * wv;
      d1 += s_fc[1][k] * wv;
    }
    float bb = bf1[f], w2v = Wf2[f];
    d0 += bb; d1 += bb;
    d0 = (d0 >= 0.f) ? d0 : SLOPE * d0;
    d1 = (d1 >= 0.f) ? d1 : SLOPE * d1;
    qp0 += d0 * w2v;
    qp1 += d1 * w2v;
  }
  atomicAdd(&s_q[0], qp0);
  atomicAdd(&s_q[1], qp1);
  __syncthreads();
  if (t < 2) out[blk * 2 + t] = s_q[t] + bf2[0];
}

// ---------------------------------------------------------------------------
extern "C" void kernel_launch(void* const* d_in, const int* in_sizes, int n_in,
                              void* d_out, int out_size, void* d_ws, size_t ws_size,
                              hipStream_t stream) {
  const float* x_feat = (const float*)d_in[0];
  const float* W1  = (const float*)d_in[1];
  const float* b1  = (const float*)d_in[2];
  const float* g1  = (const float*)d_in[3];
  const float* be1 = (const float*)d_in[4];
  const float* W2  = (const float*)d_in[5];
  const float* b2  = (const float*)d_in[6];
  const float* g2  = (const float*)d_in[7];
  const float* be2 = (const float*)d_in[8];
  const float* W3  = (const float*)d_in[9];
  const float* b3  = (const float*)d_in[10];
  const float* Wf1 = (const float*)d_in[11];
  const float* bf1 = (const float*)d_in[12];
  const float* Wf2 = (const float*)d_in[13];
  const float* bf2 = (const float*)d_in[14];
  const int* actions = (const int*)d_in[15];
  _Float16* wsH = (_Float16*)d_ws;
  float* outp = (float*)d_out;

  prep_weights<<<(WS_HALVES + 255) / 256, 256, 0, stream>>>(W1, W2, W3, wsH);
  qnet_kernel<<<BATCH / 2, 64, 0, stream>>>(x_feat, b1, g1, be1, b2, g2, be2,
                                            b3, Wf1, bf1, Wf2, bf2, actions,
                                            wsH, outp);
}
